// GeometryMlpDecoder_62105227100388
// MI455X (gfx1250) — matrix operand order repaired and checked by reading, not compiled, not run on hardware
//
#include <hip/hip_runtime.h>
#include <hip/hip_bf16.h>
#include <math.h>

typedef __attribute__((ext_vector_type(16))) _Float16 v16h;
typedef __attribute__((ext_vector_type(8)))  _Float16 v8h;
typedef __attribute__((ext_vector_type(2)))  _Float16 h2;
typedef __attribute__((ext_vector_type(8)))  float    v8f;

#define WAVES_PER_BLOCK 8

// Hash-grid level constants (match reference: b = 128^(1/15), 16 levels, 2^19 hash cap)
__constant__ int c_res[16] = {16,23,31,43,59,81,112,154,213,295,407,562,777,1073,1483,2048};
__constant__ int c_off[16] = {0,4096,16264,46056,125568,330952,855240,1379528,
                              1903816,2428104,2952392,3476680,4000968,4525256,5049544,5573832};
// levels >= 5 are hashed with size == 2^19  ->  "% size" == "& 0x7FFFF"

// Contiguous 32-byte fragment load from LDS (compiler emits 2x ds_load_b128)
__device__ __forceinline__ v16h ld_frag32(const _Float16* p) {
  const v8h lo = *(const v8h*)p;
  const v8h hi = *(const v8h*)(p + 8);
  return __builtin_shufflevector(lo, hi, 0,1,2,3,4,5,6,7,8,9,10,11,12,13,14,15);
}

// Pre-swizzle a row-major [K][outDim] f32 weight matrix into WMMA A-fragment layout:
// dst[(mt*nK + kc)*512 + lane*16 + e] holds W^T tile element A[m=lane&15][k] with
// k = 16*half + e (contiguous K order, matching the B fragments). Contiguous per lane.
__device__ __forceinline__ void swizzle_weights(const float* __restrict__ W, _Float16* dst,
                                                int nK, int nM, int ld, int validOut) {
  const int total = nM * nK * 512;
  for (int i = threadIdx.x; i < total; i += 256) {
    const int f    = i >> 9;
    const int lane = (i >> 4) & 31;
    const int e    = i & 15;
    const int mt   = f / nK;
    const int kc   = f - mt * nK;
    const int m    = lane & 15;
    const int hlf  = lane >> 4;
    const int v    = e >> 1;
    const int bit  = e & 1;
    const int kb   = (v < 4) ? (2 * v) : (16 + 2 * (v - 4));
    const int k    = kc * 32 + 16 * hlf + e;
    const int o    = mt * 16 + m;
    dst[i] = (_Float16)((o < validOut) ? W[k * ld + o] : 0.0f);
  }
}

__global__ __launch_bounds__(256) void
ngp_hashmlp_kernel(const float* __restrict__ pts, const float* __restrict__ table,
                   const float* __restrict__ Win, const float* __restrict__ Wh1,
                   const float* __restrict__ Wh2, const float* __restrict__ Wout,
                   float* __restrict__ out, int P, int numTiles)
{
  // Pre-swizzled weight fragments (A operand = W^T tiles)
  __shared__ __align__(32) _Float16 sWinF [4 * 512];   // mt 0..3, kc = 0        (32 -> 64)
  __shared__ __align__(32) _Float16 sWh1F [8 * 512];   // (mt*2 + kc)            (64 -> 64)
  __shared__ __align__(32) _Float16 sWh2F [8 * 512];   // (mt*2 + kc)            (64 -> 64)
  __shared__ __align__(32) _Float16 sWoutF[2 * 512];   // kc 0..1, mt = 0 padded (64 -> 4)
  // Per-wave activation staging, row-major [point][feat]
  __shared__ __align__(32) _Float16 sAct[WAVES_PER_BLOCK][16 * 64];

  swizzle_weights(Win,  sWinF,  1, 4, 64, 64);
  swizzle_weights(Wh1,  sWh1F,  2, 4, 64, 64);
  swizzle_weights(Wh2,  sWh2F,  2, 4, 64, 64);
  swizzle_weights(Wout, sWoutF, 2, 1,  4,  4);
  __syncthreads();

  const int tid  = threadIdx.x;
  const int wave = tid >> 5;
  const int lane = tid & 31;
  const int tile = (int)blockIdx.x * WAVES_PER_BLOCK + wave;
  if (tile >= numTiles) return;   // wave-uniform: EXEC stays all-1s for WMMA

  _Float16* stage = &sAct[wave][0];
  const int m   = lane & 15;      // point within tile (B-operand N index)
  const int hlf = lane >> 4;

  // ------------- hash-grid encode: 16 points x 32 f16 features, [point][feat] -------------
  {
    int p = tile * 16 + m;
    if (p >= P) p = P - 1;
    const float px = pts[3 * p + 0];
    const float py = pts[3 * p + 1];
    const float pz = pts[3 * p + 2];
    const float2* tab = (const float2*)table;
    const int lvl0 = hlf * 8;     // lanes 0-15: levels 0-7; lanes 16-31: levels 8-15
#pragma unroll
    for (int li = 0; li < 8; ++li) {
      const int level = lvl0 + li;
      const int res   = c_res[level];
      const int off   = c_off[level];
      const float rf  = (float)(res - 1);
      const float fx = px * rf, fy = py * rf, fz = pz * rf;
      int x0 = (int)floorf(fx); x0 = min(max(x0, 0), res - 2);
      int y0 = (int)floorf(fy); y0 = min(max(y0, 0), res - 2);
      int z0 = (int)floorf(fz); z0 = min(max(z0, 0), res - 2);
      const float tx = fx - (float)x0, ty = fy - (float)y0, tz = fz - (float)z0;

      // Phase A: compute all 8 corner indices and ISSUE all 8 gathers (batched clause,
      // one descending s_wait_loadcnt -> 8-deep memory-level parallelism per level)
      float2 tv[8];
#pragma unroll
      for (int c = 0; c < 8; ++c) {
        const unsigned cx = (unsigned)(x0 + ((c >> 2) & 1));
        const unsigned cy = (unsigned)(y0 + ((c >> 1) & 1));
        const unsigned cz = (unsigned)(z0 + (c & 1));
        int idx;
        if (level < 5) {  // dense level
          idx = (int)(cx + (unsigned)res * (cy + (unsigned)res * cz));
        } else {          // hashed level, size == 2^19
          const unsigned h = cx ^ (cy * 2654435761u) ^ (cz * 805459861u);
          idx = (int)(h & 0x7FFFFu);
        }
        tv[c] = tab[off + idx];           // L2-resident gather (48MB << 192MB L2)
      }

      // Phase B: trilinear accumulate
      const float wx[2] = { 1.0f - tx, tx };
      const float wy[2] = { 1.0f - ty, ty };
      const float wz[2] = { 1.0f - tz, tz };
      float f0 = 0.0f, f1 = 0.0f;
#pragma unroll
      for (int c = 0; c < 8; ++c) {
        const float w = wx[(c >> 2) & 1] * wy[(c >> 1) & 1] * wz[c & 1];
        f0 = fmaf(tv[c].x, w, f0);
        f1 = fmaf(tv[c].y, w, f1);
      }
      h2 fe; fe[0] = (_Float16)f0; fe[1] = (_Float16)f1;
      *(h2*)&stage[m * 32 + 2 * level] = fe;         // packed b32 store
    }
  }
  // same-wave LDS RAW below: DS ops are in-order per wave; compiler inserts s_wait_dscnt

  v8f acc[4];
  const v8f zero = {};

  // ------------- layer 1: D[64,16] = Win^T[64x32] x enc^T[32x16], ReLU -------------
  {
    const v16h b = ld_frag32(stage + m * 32 + 16 * hlf);   // enc as B: K=16*half+0..15 contiguous
#pragma unroll
    for (int mt = 0; mt < 4; ++mt) {
      const v16h a = ld_frag32(sWinF + mt * 512 + lane * 16);
      acc[mt] = __builtin_amdgcn_wmma_f32_16x16x32_f16(
          false, a, false, b, (short)0, zero, false, false);
    }
#pragma unroll
    for (int mt = 0; mt < 4; ++mt) {     // ReLU + pack 8 rows -> one b128 store
      v8h h;
#pragma unroll
      for (int r = 0; r < 8; ++r) { const float v = acc[mt][r]; h[r] = (_Float16)(v > 0.0f ? v : 0.0f); }
      *(v8h*)&stage[m * 64 + mt * 16 + 8 * hlf] = h;
    }
  }

  // ------------- layers 2 & 3: D[64,16] = W^T[64x64] x act^T[64x16], ReLU -------------
  const _Float16* Whid[2] = { sWh1F, sWh2F };
#pragma unroll
  for (int layer = 0; layer < 2; ++layer) {
    const _Float16* WF = Whid[layer];
    const v16h b0 = ld_frag32(stage + m * 64 +      16 * hlf);   // K-chunk 0
    const v16h b1 = ld_frag32(stage + m * 64 + 32 + 16 * hlf);   // K-chunk 1
#pragma unroll
    for (int mt = 0; mt < 4; ++mt) {
      const v8f c = __builtin_amdgcn_wmma_f32_16x16x32_f16(
          false, ld_frag32(WF + (mt * 2 + 0) * 512 + lane * 16), false, b0, (short)0, zero, false, false);
      acc[mt] = __builtin_amdgcn_wmma_f32_16x16x32_f16(
          false, ld_frag32(WF + (mt * 2 + 1) * 512 + lane * 16), false, b1, (short)0, c, false, false);
    }
#pragma unroll
    for (int mt = 0; mt < 4; ++mt) {
      v8h h;
#pragma unroll
      for (int r = 0; r < 8; ++r) { const float v = acc[mt][r]; h[r] = (_Float16)(v > 0.0f ? v : 0.0f); }
      *(v8h*)&stage[m * 64 + mt * 16 + 8 * hlf] = h;
    }
  }

  // ------------- output layer: D[16(pad),16] = Wout^T x act^T; no LDS epilogue -------------
  {
    const v16h b0 = ld_frag32(stage + m * 64 +      16 * hlf);
    const v16h b1 = ld_frag32(stage + m * 64 + 32 + 16 * hlf);
    v8f o = __builtin_amdgcn_wmma_f32_16x16x32_f16(
        false, ld_frag32(sWoutF + 0 * 512 + lane * 16), false, b0, (short)0, zero, false, false);
    o = __builtin_amdgcn_wmma_f32_16x16x32_f16(
        false, ld_frag32(sWoutF + 1 * 512 + lane * 16), false, b1, (short)0, o, false, false);

    // D rows = outdim (r + 8*half), cols = point m: lanes 0-15 hold outs 0..3 of point=lane
    if (lane < 16) {
      const int p = tile * 16 + lane;
      if (p < P) {
        const float o0 = o[0], o1 = o[1], o2 = o[2], o3 = o[3];
        const float density = (o0 > 20.0f) ? o0 : log1pf(expf(o0));
        const float t0 = tanhf(o1), t1 = tanhf(o2), t2 = tanhf(o3);
        float nrm = sqrtf(t0 * t0 + t1 * t1 + t2 * t2);
        nrm = fmaxf(nrm, 1e-12f);
        const float inv = 1.0f / nrm;
        out[p] = density;                  // density [P]
        float* onrm = out + P;             // normal [P][3], concatenated after density
        onrm[3 * p + 0] = t0 * inv;
        onrm[3 * p + 1] = t1 * inv;
        onrm[3 * p + 2] = t2 * inv;
      }
    }
  }
}

extern "C" void kernel_launch(void* const* d_in, const int* in_sizes, int n_in,
                              void* d_out, int out_size, void* d_ws, size_t ws_size,
                              hipStream_t stream) {
  const float* pts   = (const float*)d_in[0];
  const float* table = (const float*)d_in[1];
  const float* Win   = (const float*)d_in[2];
  const float* Wh1   = (const float*)d_in[3];
  const float* Wh2   = (const float*)d_in[4];
  const float* Wout  = (const float*)d_in[5];
  float* out = (float*)d_out;

  const int P = in_sizes[0] / 3;                 // 524288 points
  const int numTiles = (P + 15) / 16;            // 16-point WMMA tiles
  const int blocks = (numTiles + WAVES_PER_BLOCK - 1) / WAVES_PER_BLOCK;
  hipLaunchKernelGGL(ngp_hashmlp_kernel, dim3(blocks), dim3(256), 0, stream,
                     pts, table, Win, Wh1, Wh2, Wout, out, P, numTiles);
}
